// Conv1d_NN_Spatial_44976897523805
// MI455X (gfx1250) — matrix operand order repaired and checked by reading, not compiled, not run on hardware
//
#include <hip/hip_runtime.h>
#include <hip/hip_bf16.h>

// Problem dims (fixed by the reference setup)
#define B_   32
#define CIN  64
#define COUT 128
#define N_   4096
#define M_   1024
#define KNN  9          // neighbors incl. self
#define KDIM (CIN*KNN)  // 576 = conv GEMM K-dim

typedef __attribute__((ext_vector_type(2)))  float  v2f;
typedef __attribute__((ext_vector_type(8)))  float  v8f;
typedef __attribute__((ext_vector_type(16))) __bf16 v16bf;

// ---------------------------------------------------------------------------
// Kernel 1: ny[b,m] = sum_c y[b,c,m]^2  (nx dropped: constant per row; sqrt
// dropped: monotonic — neither affects per-row top-k ordering)
// ---------------------------------------------------------------------------
__global__ __launch_bounds__(256) void ny_kernel(const float* __restrict__ y,
                                                 float* __restrict__ ny) {
  int gid = blockIdx.x * blockDim.x + threadIdx.x;
  if (gid >= B_ * M_) return;
  int b = gid / M_, m = gid - b * M_;
  const float* yp = y + (size_t)b * CIN * M_ + m;
  float s = 0.f;
#pragma unroll 8
  for (int c = 0; c < CIN; ++c) {
    float v = yp[c * M_];
    s += v * v;
  }
  ny[gid] = s;
}

// ---------------------------------------------------------------------------
// Kernel 2: transpose-convert y -> ybf[b][m][c] (bf16). Makes every WMMA
// B-fragment in the distance GEMM one contiguous 32-byte per-lane load.
// LDS-tiled: coalesced f32 reads along m, coalesced bf16 writes along c.
// ---------------------------------------------------------------------------
__global__ __launch_bounds__(256) void ybf_kernel(const float* __restrict__ y,
                                                  __bf16* __restrict__ ybf) {
  __shared__ float tile[CIN][65];            // 64x64 tile, padded
  const int b  = blockIdx.x >> 4;            // M_/64 = 16 m-tiles per batch
  const int m0 = (blockIdx.x & 15) << 6;
  const int t  = threadIdx.x;
  const float* yb = y + (size_t)b * CIN * M_;
  for (int i = t; i < CIN * 64; i += 256) {
    int c = i >> 6, m = i & 63;
    tile[c][m] = yb[c * M_ + m0 + m];        // coalesced along m
  }
  __syncthreads();
  __bf16* ob = ybf + ((size_t)b * M_ + m0) * CIN;
  for (int i = t; i < 64 * CIN; i += 256) {
    int m = i >> 6, c = i & 63;
    ob[m * CIN + c] = (__bf16)tile[c][m];    // coalesced along c
  }
}

// ---------------------------------------------------------------------------
// Kernel 3: W swizzle: Wtp[kb/4][g][co][p] = W[co][kb + 2g + p]  (once).
// Each conv WMMA A-fragment becomes one coalesced b64 load per lane.
// Wtp = 288 KB -> L2 resident.
// ---------------------------------------------------------------------------
__global__ __launch_bounds__(256) void wtz_kernel(const float* __restrict__ W,
                                                  float* __restrict__ Wtp) {
  int gid = blockIdx.x * blockDim.x + threadIdx.x;   // co*KDIM + kk (coalesced read)
  if (gid >= COUT * KDIM) return;
  int co = gid / KDIM, kk = gid - co * KDIM;
  int grp = ((kk >> 2) << 1) + ((kk >> 1) & 1);      // (kb/4)*2 + g
  Wtp[(size_t)grp * (COUT * 2) + co * 2 + (kk & 1)] = W[gid];
}

// ---------------------------------------------------------------------------
// Kernel 4: fused distance-GEMM (bf16 WMMA 16x16x32, f32 accumulate) +
// per-row top-8 selection. One wave32 per (batch, 32-row n-tile): two 16-row
// WMMA tiles share each y-fragment; all 32 lanes own one row each in the
// ordered-insertion scan (matches lax.top_k ascending order, lower-index ties).
// bf16 is ranking-only precision: values never reach the output.
// ---------------------------------------------------------------------------
__global__ __launch_bounds__(32) void topk_kernel(const float* __restrict__ x,
                                                  const __bf16* __restrict__ ybf,
                                                  const int*   __restrict__ indices,
                                                  const float* __restrict__ ny,
                                                  int*         __restrict__ fout) {
  __shared__ float dtile[32][17];            // +1 pad vs bank conflicts
  const int l   = threadIdx.x;
  const int b   = blockIdx.x >> 7;           // N_/32 = 128 tiles per batch
  const int n0  = (blockIdx.x & 127) << 5;   // 32 rows per workgroup
  const int col = l & 15;
  const int g   = l >> 4;                    // lane half (ISA frag K grouping)

  // Preload + convert two A panels (rows n0..n0+15 and n0+16..n0+31).
  // A K order chosen here (B uses the same): lane's K set = g*16 + {0..15} (+32 for step 1)
  v16bf a0[2], a1[2];
  const float* xb = x + (size_t)b * CIN * N_;
#pragma unroll
  for (int s = 0; s < 2; ++s) {
#pragma unroll
    for (int j = 0; j < 16; ++j) {
      int c = s * 32 + g * 16 + j;
      a0[s][j] = (__bf16)xb[c * N_ + n0 + col];
      a1[s][j] = (__bf16)xb[c * N_ + n0 + 16 + col];
    }
  }

  float bd[8]; int bm[8];
#pragma unroll
  for (int j = 0; j < 8; ++j) { bd[j] = __builtin_inff(); bm[j] = 0; }

  const float* nyb = ny + b * M_;

  for (int m0 = 0; m0 < M_; m0 += 16) {
    // B-fragment lane K set = g*16 + {0..15}: contiguous in ybf[b][m][*]
    const __bf16* ybm = ybf + ((size_t)b * M_ + m0 + col) * CIN + g * 16;
    v8f acc0 = {}, acc1 = {};
#pragma unroll
    for (int s = 0; s < 2; ++s) {
      v16bf bfr = *(const v16bf*)(ybm + s * 32);   // 2x global_load_b128
      acc0 = __builtin_amdgcn_wmma_f32_16x16x32_bf16(false, a0[s], false, bfr,
                                                     (short)0, acc0, false, false);
      acc1 = __builtin_amdgcn_wmma_f32_16x16x32_bf16(false, a1[s], false, bfr,
                                                     (short)0, acc1, false, false);
    }
    // dist' = ny[m] - 2*dot  (ranking-equivalent to full distance)
    float nyv = nyb[m0 + col];
#pragma unroll
    for (int r = 0; r < 8; ++r) {
      int row = g ? (r + 8) : r;
      dtile[row][col]      = nyv - 2.0f * acc0[r];
      dtile[16 + row][col] = nyv - 2.0f * acc1[r];
    }
    asm volatile("s_wait_dscnt 0" ::: "memory");   // tile visible across lanes
    // all 32 lanes: ordered insertion scan over this tile's 16 columns
    for (int c = 0; c < 16; ++c) {
      float d = dtile[l][c];
      if (d < bd[7]) {                 // strict < keeps earliest m on ties
        float v = d; int mi = m0 + c;
#pragma unroll
        for (int j = 0; j < 8; ++j) {
          if (v < bd[j]) {
            float td = bd[j]; int tm = bm[j];
            bd[j] = v; bm[j] = mi; v = td; mi = tm;
          }
        }
      }
    }
    asm volatile("s_wait_dscnt 0" ::: "memory");   // reads done before overwrite
  }

  int n = n0 + l;
  int* op = fout + ((size_t)b * N_ + n) * KNN;
  op[0] = n;                                       // self index
#pragma unroll
  for (int j = 0; j < 8; ++j) op[1 + j] = indices[bm[j]];
}

// ---------------------------------------------------------------------------
// Kernel 5: gather + conv(k=9, stride=9) as full-precision f32 GEMM:
//   out[b, co, n] = sum_{kk=ci*9+k} W[co, kk] * x[b, ci, fidx[b,n,k]] + bias[co]
// One wave32 per (batch, 16-n tile). Gathered A-panel (576 x 16 f32 = 36 KB)
// staged in LDS (pair-swizzled -> ds_load_b64); 8 co-tiles x 144 chained K=4
// f32 WMMAs; W fragments are coalesced b64 loads from Wtp; stores coalesced
// along n.
// ---------------------------------------------------------------------------
__global__ __launch_bounds__(32) void conv_kernel(const float* __restrict__ x,
                                                  const float* __restrict__ Wtp,
                                                  const float* __restrict__ bias,
                                                  const int*   __restrict__ fidx,
                                                  float*       __restrict__ out) {
  __shared__ __align__(16) float Ap[KDIM * 16];   // pair-swizzled [grp][ncol][2]
  __shared__ int   fi[16 * KNN];
  __shared__ float bs[COUT];
  const int l   = threadIdx.x;
  const int b   = blockIdx.x >> 8;
  const int n0  = (blockIdx.x & 255) << 4;
  const int col = l & 15;
  const int g   = l >> 4;              // f32 frag: K pair (2g, 2g+1) per half

  for (int i = l; i < 16 * KNN; i += 32)
    fi[i] = fidx[((size_t)b * N_ + n0 + i / KNN) * KNN + (i % KNN)];
  for (int i = l; i < COUT; i += 32) bs[i] = bias[i];
  asm volatile("s_wait_dscnt 0" ::: "memory");

  // Stage gathered neighbor-feature panel into LDS (pair-swizzled layout).
  const float* xb = x + (size_t)b * CIN * N_;
  for (int i = l; i < KDIM * 16; i += 32) {
    int ncol = i & 15;
    int kk   = i >> 4;
    int ci   = kk / KNN;
    int k    = kk - ci * KNN;
    int sidx = fi[ncol * KNN + k];
    int grp  = ((kk >> 2) << 1) + ((kk >> 1) & 1);
    Ap[(grp << 5) + (ncol << 1) + (kk & 1)] = xb[ci * N_ + sidx];
  }
  asm volatile("s_wait_dscnt 0" ::: "memory");

  float* ob = out + (size_t)b * COUT * N_;
  for (int t = 0; t < 8; ++t) {
    const int co0 = t << 4;
    v8f acc = {};
#pragma unroll 4
    for (int kb = 0; kb < KDIM; kb += 4) {
      const int grp = ((kb >> 2) << 1) + g;
      // A frag: W rows (co): one coalesced b64 load per lane
      v2f a = *(const v2f*)(Wtp + (size_t)grp * (COUT * 2) + ((co0 + col) << 1));
      // B frag: gathered panel columns (n): one ds_load_b64 per lane
      v2f bb = *(const v2f*)(Ap + (grp << 5) + (col << 1));
      acc = __builtin_amdgcn_wmma_f32_16x16x4_f32(false, a, false, bb,
                                                  (short)0, acc, false, false);
    }
#pragma unroll
    for (int r = 0; r < 8; ++r) {
      int co = co0 + (g ? (r + 8) : r);
      ob[(size_t)co * N_ + n0 + col] = acc[r] + bs[co];   // coalesced along n
    }
  }
}

// ---------------------------------------------------------------------------
extern "C" void kernel_launch(void* const* d_in, const int* in_sizes, int n_in,
                              void* d_out, int out_size, void* d_ws, size_t ws_size,
                              hipStream_t stream) {
  const float* x       = (const float*)d_in[0];
  const float* y       = (const float*)d_in[1];
  const int*   indices = (const int*)  d_in[2];
  const float* W       = (const float*)d_in[3];
  const float* bias    = (const float*)d_in[4];
  float* out = (float*)d_out;

  // Workspace: ny [B*M] f32 | fidx [B*N*9] i32 | Wtp [576*128] f32 | ybf [B*M*64] bf16
  char* ws = (char*)d_ws;
  size_t off = 0;
  float* ny = (float*)(ws + off);
  off += ((size_t)B_ * M_ * sizeof(float) + 255) & ~(size_t)255;
  int* fidx = (int*)(ws + off);
  off += ((size_t)B_ * N_ * KNN * sizeof(int) + 255) & ~(size_t)255;
  float* Wtp = (float*)(ws + off);
  off += ((size_t)KDIM * COUT * sizeof(float) + 255) & ~(size_t)255;
  __bf16* ybf = (__bf16*)(ws + off);

  hipLaunchKernelGGL(ny_kernel, dim3((B_ * M_ + 255) / 256), dim3(256), 0, stream,
                     y, ny);
  hipLaunchKernelGGL(ybf_kernel, dim3(B_ * (M_ / 64)), dim3(256), 0, stream,
                     y, ybf);
  hipLaunchKernelGGL(wtz_kernel, dim3((COUT * KDIM + 255) / 256), dim3(256), 0, stream,
                     W, Wtp);
  hipLaunchKernelGGL(topk_kernel, dim3(B_ * (N_ / 32)), dim3(32), 0, stream,
                     x, ybf, indices, ny, fidx);
  hipLaunchKernelGGL(conv_kernel, dim3(B_ * (N_ / 16)), dim3(32), 0, stream,
                     x, Wtp, bias, fidx, out);
}
